// LSTMModel_31679678775668
// MI455X (gfx1250) — matrix operand order repaired and checked by reading, not compiled, not run on hardware
//
#include <hip/hip_runtime.h>
#include <hip/hip_bf16.h>
#include <stdint.h>

// ---------------------------------------------------------------------------
// Problem constants
// ---------------------------------------------------------------------------
#define Tn 512
#define Bn 256
#define Dn 64
#define Hn 128
#define NOUT 1536            // LEN_OUT * D = 24*64
#define K1 (Dn + Hn)         // 192
#define K2 (Hn + Hn)         // 256
#define G4H (4 * Hn)         // 512
#define HP (Hn + 8)          // padded LDS stride for h-state (bank-conflict-free)
#define K2P (K2 + 8)         // padded LDS stride for W2 tile (bank-conflict-free)

typedef __attribute__((ext_vector_type(16))) __bf16 bf16x16;
typedef __attribute__((ext_vector_type(8)))  __bf16 bf16x8;
typedef __attribute__((ext_vector_type(8)))  float  f32x8;

// ---------------------------------------------------------------------------
// Fragment loaders (CDNA5 wave32 WMMA layouts, CDNA5 ISA §7.12.2)
// A (16x32 bf16, row-major source, leading dim = ld):
//   lane<16: M=lane,    holds K = 0..7   (v0..v3) and 16..23 (v4..v7)
//   lane>=16:M=lane-16, holds K = 8..15  and 24..31
// ---------------------------------------------------------------------------
__device__ __forceinline__ bf16x16 load_a(const __bf16* p, int ld, int lane) {
  const __bf16* q = p + (lane & 15) * ld + ((lane >> 4) << 3);
  bf16x8 lo = *(const bf16x8*)(q);        // K kb .. kb+7
  bf16x8 hi = *(const bf16x8*)(q + 16);   // K kb+16 .. kb+23
  return __builtin_shufflevector(lo, hi, 0,1,2,3,4,5,6,7,8,9,10,11,12,13,14,15);
}

// B (32x16 bf16). Source is the TRANSPOSED weight matrix [N,K] row-major so a
// lane's K-values match A: lane<16 col=lane K=0..7,16..23, lane>=16 K=8..15,24..31.
__device__ __forceinline__ bf16x16 load_b(const __bf16* wT, int ld, int lane) {
  const __bf16* q = wT + (lane & 15) * ld + ((lane >> 4) << 3);
  bf16x8 lo = *(const bf16x8*)(q);        // K kb .. kb+7
  bf16x8 hi = *(const bf16x8*)(q + 16);   // K kb+16 .. kb+23
  return __builtin_shufflevector(lo, hi, 0,1,2,3,4,5,6,7,8,9,10,11,12,13,14,15);
}

__device__ __forceinline__ f32x8 wmma_bf16(bf16x16 a, bf16x16 b, f32x8 c) {
  return __builtin_amdgcn_wmma_f32_16x16x32_bf16(
      false, a, false, b, (short)0, c, false, false);
}

__device__ __forceinline__ float sigm(float x) {
  return 1.0f / (1.0f + __expf(-x));
}
__device__ __forceinline__ float tanh_f(float x) {
  float e = __expf(2.0f * fabsf(x));          // inf-safe
  return copysignf(1.0f - 2.0f / (e + 1.0f), x);
}

// ---------------------------------------------------------------------------
// Prep kernels: bf16 transpose-convert of weights, time-major bf16 features
// ---------------------------------------------------------------------------
__global__ void k_prep_weights(const float* __restrict__ W1,
                               const float* __restrict__ W2,
                               const float* __restrict__ Wd,
                               __bf16* __restrict__ w1t,
                               __bf16* __restrict__ w2t,
                               __bf16* __restrict__ wdt,
                               unsigned* done) {
  int i = blockIdx.x * blockDim.x + threadIdx.x;
  if (i == 0) *done = 0u;
  if (i < G4H * K1) {                       // w1t[n][k] = W1[k][n]
    int n = i / K1, k = i % K1;
    w1t[i] = (__bf16)W1[k * G4H + n];
    return;
  }
  int j = i - G4H * K1;
  if (j < G4H * K2) {                       // w2t[n][k] = W2[k][n]
    int n = j / K2, k = j % K2;
    w2t[j] = (__bf16)W2[k * G4H + n];
    return;
  }
  int l = j - G4H * K2;
  if (l < NOUT * Hn) {                      // wdt[n][k] = Wd[k][n]
    int n = l / Hn, k = l % Hn;
    wdt[l] = (__bf16)Wd[k * NOUT + n];
  }
}

__global__ void k_prep_x(const float* __restrict__ feat, __bf16* __restrict__ xbf) {
  int i = blockIdx.x * blockDim.x + threadIdx.x;   // over T*B*D
  if (i >= Tn * Bn * Dn) return;
  int d = i % Dn;
  int b = (i / Dn) % Bn;
  int t = i / (Dn * Bn);
  xbf[i] = (__bf16)feat[(size_t)b * Tn * Dn + (size_t)t * Dn + d];  // [T,B,D]
}

// ---------------------------------------------------------------------------
// Persistent fused 2-layer LSTM + dense + loss.
// Grid = 16 blocks (one per 16 batch rows) x 256 threads (8 waves).
// Wave w owns h-unit tile [16w, 16w+16); recurrence is block-local (LDS).
// W1^T fragments live in VGPRs (192 regs); W2^T lives in LDS (264 KB, padded).
// ---------------------------------------------------------------------------
__launch_bounds__(256, 1)
__global__ void k_lstm(const __bf16* __restrict__ w1t, const __bf16* __restrict__ w2t,
                       const __bf16* __restrict__ wdt, const __bf16* __restrict__ xbf,
                       const float* __restrict__ b1,  const float* __restrict__ b2,
                       const float* __restrict__ bd,  const float* __restrict__ labels,
                       float* __restrict__ out, float* blockpart, unsigned* done) {
  __shared__ alignas(16) __bf16 w2s[G4H * K2P];   // 512 x 264 bf16 = 264 KB
  __shared__ alignas(16) __bf16 h1s[16 * HP];     // 16 x 136
  __shared__ alignas(16) __bf16 h2s[16 * HP];
  __shared__ float wsum[8];

  const int lane  = threadIdx.x & 31;
  const int wv    = threadIdx.x >> 5;         // h-tile 0..7
  const int mtile = blockIdx.x;               // 0..15 (batch rows mtile*16..+15)
  const int col   = lane & 15;
  const int rbase = (lane >> 4) << 3;

  // cooperative copy W2^T (row-major [512,256]) into padded LDS [512,264]
  for (int idx = threadIdx.x; idx < G4H * (K2 / 16); idx += 256) {
    int n = idx / (K2 / 16), s = (idx % (K2 / 16)) * 16;
    *(bf16x8*)(&w2s[n * K2P + s])     = *(const bf16x8*)(w2t + n * K2 + s);
    *(bf16x8*)(&w2s[n * K2P + s + 8]) = *(const bf16x8*)(w2t + n * K2 + s + 8);
  }
  for (int i = threadIdx.x; i < 16 * HP; i += 256) {
    h1s[i] = (__bf16)0.0f;
    h2s[i] = (__bf16)0.0f;
  }
  __syncthreads();

  // preload layer-1 B fragments into registers (24 frags = 192 VGPRs)
  bf16x16 w1x[2][4];                          // x-part, K chunks 0..1
  bf16x16 w1h[4][4];                          // h-part, K chunks 0..3
#pragma unroll
  for (int kc = 0; kc < 2; ++kc)
#pragma unroll
    for (int g = 0; g < 4; ++g)
      w1x[kc][g] = load_b(w1t + (size_t)(g * Hn + wv * 16) * K1 + kc * 32, K1, lane);
#pragma unroll
  for (int kc = 0; kc < 4; ++kc)
#pragma unroll
    for (int g = 0; g < 4; ++g)
      w1h[kc][g] = load_b(w1t + (size_t)(g * Hn + wv * 16) * K1 + Dn + kc * 32, K1, lane);

  float c1[8], c2[8];
#pragma unroll
  for (int r = 0; r < 8; ++r) { c1[r] = 0.0f; c2[r] = 0.0f; }

  float b1g[4], b2g[4];
#pragma unroll
  for (int g = 0; g < 4; ++g) {
    b1g[g] = b1[g * Hn + wv * 16 + col];
    b2g[g] = b2[g * Hn + wv * 16 + col];
  }

#pragma unroll 1
  for (int t = 0; t < Tn; ++t) {
    // -------- layer 1: z = [x_t | h1] @ W1 + b1 --------------------------
    const __bf16* xp = xbf + ((size_t)t * Bn + mtile * 16) * Dn;
    // prefetch next step's x tile (32 lanes x 64B == 2KB tile)
    __builtin_prefetch((const char*)(xbf + ((size_t)((t + 1) & (Tn - 1)) * Bn
                                            + mtile * 16) * Dn) + lane * 64, 0, 3);
    f32x8 acc[4] = {};
#pragma unroll
    for (int kc = 0; kc < 2; ++kc) {                       // x part, K=64
      bf16x16 a = load_a(xp + kc * 32, Dn, lane);
#pragma unroll
      for (int g = 0; g < 4; ++g)
        acc[g] = wmma_bf16(a, w1x[kc][g], acc[g]);
    }
#pragma unroll
    for (int kc = 0; kc < 4; ++kc) {                       // h1 part, K=128
      bf16x16 a = load_a(h1s + kc * 32, HP, lane);
#pragma unroll
      for (int g = 0; g < 4; ++g)
        acc[g] = wmma_bf16(a, w1h[kc][g], acc[g]);
    }
    __syncthreads();                                       // all reads of h1s(t-1) done
#pragma unroll
    for (int r = 0; r < 8; ++r) {                          // TF gate order i,j,f,o
      float zi = acc[0][r] + b1g[0];
      float zj = acc[1][r] + b1g[1];
      float zf = acc[2][r] + b1g[2];
      float zo = acc[3][r] + b1g[3];
      float c  = c1[r] * sigm(zf + 1.0f) + sigm(zi) * tanh_f(zj);
      c1[r] = c;
      h1s[(rbase + r) * HP + wv * 16 + col] = (__bf16)(tanh_f(c) * sigm(zo));
    }
    __syncthreads();                                       // h1s(t) visible

    // -------- layer 2: z = [h1_t | h2] @ W2 + b2 (B frags stream from LDS)
#pragma unroll
    for (int g = 0; g < 4; ++g) acc[g] = (f32x8){};        // reuse accumulators
#pragma unroll
    for (int kc = 0; kc < 4; ++kc) {                       // h1 part
      bf16x16 a = load_a(h1s + kc * 32, HP, lane);
#pragma unroll
      for (int g = 0; g < 4; ++g) {
        bf16x16 bm = load_b(&w2s[(g * Hn + wv * 16) * K2P + kc * 32], K2P, lane);
        acc[g] = wmma_bf16(a, bm, acc[g]);
      }
    }
#pragma unroll
    for (int kc = 0; kc < 4; ++kc) {                       // h2 part
      bf16x16 a = load_a(h2s + kc * 32, HP, lane);
#pragma unroll
      for (int g = 0; g < 4; ++g) {
        bf16x16 bm = load_b(&w2s[(g * Hn + wv * 16) * K2P + Hn + kc * 32], K2P, lane);
        acc[g] = wmma_bf16(a, bm, acc[g]);
      }
    }
    __syncthreads();                                       // all reads of h2s(t-1) done
#pragma unroll
    for (int r = 0; r < 8; ++r) {
      float zi = acc[0][r] + b2g[0];
      float zj = acc[1][r] + b2g[1];
      float zf = acc[2][r] + b2g[2];
      float zo = acc[3][r] + b2g[3];
      float c  = c2[r] * sigm(zf + 1.0f) + sigm(zi) * tanh_f(zj);
      c2[r] = c;
      h2s[(rbase + r) * HP + wv * 16 + col] = (__bf16)(tanh_f(c) * sigm(zo));
    }
    __syncthreads();                                       // h2s(t) visible
  }

  // -------- dense: pred = h2_last @ Wd + bd; MSE partials ----------------
  float partial = 0.0f;
  for (int j = 0; j < 12; ++j) {                           // 96 N-tiles / 8 waves
    int n0 = (wv + 8 * j) * 16;
    f32x8 accd = {};
#pragma unroll
    for (int kc = 0; kc < 4; ++kc) {
      bf16x16 a  = load_a(h2s + kc * 32, HP, lane);
      bf16x16 bm = load_b(wdt + (size_t)n0 * Hn + kc * 32, Hn, lane);
      accd = wmma_bf16(a, bm, accd);
    }
    int cidx = n0 + col;
    float bdv = bd[cidx];
#pragma unroll
    for (int r = 0; r < 8; ++r) {
      int m = mtile * 16 + rbase + r;
      float p = accd[r] + bdv;
      out[(size_t)m * NOUT + cidx] = p;
      float d = p - labels[(size_t)m * NOUT + cidx];
      partial += d * d;
    }
  }
  // deterministic reduction: lane -> wave -> block -> last-done block
#pragma unroll
  for (int off = 16; off > 0; off >>= 1)
    partial += __shfl_xor(partial, off, 32);
  if (lane == 0) wsum[wv] = partial;
  __syncthreads();
  if (threadIdx.x == 0) {
    float s = 0.0f;
    for (int i = 0; i < 8; ++i) s += wsum[i];
    __hip_atomic_store(&blockpart[mtile], s, __ATOMIC_RELEASE, __HIP_MEMORY_SCOPE_AGENT);
    unsigned old = __hip_atomic_fetch_add(done, 1u, __ATOMIC_ACQ_REL,
                                          __HIP_MEMORY_SCOPE_AGENT);
    if (old == 15u) {
      float tot = 0.0f;
      for (int b = 0; b < 16; ++b)
        tot += __hip_atomic_load(&blockpart[b], __ATOMIC_ACQUIRE,
                                 __HIP_MEMORY_SCOPE_AGENT);
      out[(size_t)Bn * NOUT] = tot / (float)(Bn * NOUT);
    }
  }
}

// ---------------------------------------------------------------------------
// kernel_launch
// ---------------------------------------------------------------------------
extern "C" void kernel_launch(void* const* d_in, const int* in_sizes, int n_in,
                              void* d_out, int out_size, void* d_ws, size_t ws_size,
                              hipStream_t stream) {
  (void)in_sizes; (void)n_in; (void)out_size; (void)ws_size;
  const float* feat   = (const float*)d_in[0];   // [B,T,D]
  const float* labels = (const float*)d_in[1];   // [B,24,D] == [B,1536]
  const float* W1     = (const float*)d_in[2];   // [192,512]
  const float* b1     = (const float*)d_in[3];   // [512]
  const float* W2     = (const float*)d_in[4];   // [256,512]
  const float* b2     = (const float*)d_in[5];   // [512]
  const float* Wd     = (const float*)d_in[6];   // [128,1536]
  const float* bd     = (const float*)d_in[7];   // [1536]
  float* out = (float*)d_out;                    // pred [256,1536] ++ loss[1]

  uintptr_t base = (uintptr_t)d_ws;
  auto alloc = [&](size_t bytes) -> void* {
    base = (base + 255) & ~(uintptr_t)255;
    void* p = (void*)base;
    base += bytes;
    return p;
  };
  __bf16*   w1t = (__bf16*)alloc((size_t)G4H * K1 * 2);        // [512,192]
  __bf16*   w2t = (__bf16*)alloc((size_t)G4H * K2 * 2);        // [512,256]
  __bf16*   wdt = (__bf16*)alloc((size_t)NOUT * Hn * 2);       // [1536,128]
  __bf16*   xbf = (__bf16*)alloc((size_t)Tn * Bn * Dn * 2);    // [T,B,D]
  float* blockpart = (float*)alloc(16 * sizeof(float));
  unsigned* done   = (unsigned*)alloc(sizeof(unsigned));

  const int wtot = G4H * K1 + G4H * K2 + NOUT * Hn;            // 425984
  k_prep_weights<<<(wtot + 255) / 256, 256, 0, stream>>>(W1, W2, Wd, w1t, w2t, wdt, done);
  const int xtot = Tn * Bn * Dn;                               // 8388608
  k_prep_x<<<(xtot + 255) / 256, 256, 0, stream>>>(feat, xbf);
  k_lstm<<<16, 256, 0, stream>>>(w1t, w2t, wdt, xbf, b1, b2, bd, labels,
                                 out, blockpart, done);
}
